// MolCLRGinLayer_67353677136443
// MI455X (gfx1250) — compile-verified, matrix operand order checked by reading, not run on hardware
//
#include <hip/hip_runtime.h>
#include <hip/hip_bf16.h>

#define N_NODES 50000
#define N_EDGES 600000
#define DIM     128
#define DIM2    256

typedef __attribute__((ext_vector_type(16))) __bf16 v16bf;
typedef __attribute__((ext_vector_type(8)))  __bf16 v8bf;
typedef __attribute__((ext_vector_type(8)))  float  v8f;

__device__ __forceinline__ unsigned short f2bf(float f) {
    unsigned u = __float_as_uint(f);
    u += 0x7FFFu + ((u >> 16) & 1u);   // round-to-nearest-even
    return (unsigned short)(u >> 16);
}

// ---------------------------------------------------------------------------
// K0: out32[i] = (1 + eps) * x[i]   (residual pre-init, scatter adds into it)
// ---------------------------------------------------------------------------
__global__ void init_out_kernel(const float* __restrict__ x,
                                const float* __restrict__ eps,
                                float* __restrict__ out32) {
    int i = blockIdx.x * blockDim.x + threadIdx.x;      // float4 index
    float s = 1.0f + eps[0];
    float4 v = ((const float4*)x)[i];
    float4 r = {v.x * s, v.y * s, v.z * s, v.w * s};
    ((float4*)out32)[i] = r;
}

// ---------------------------------------------------------------------------
// K1: convert W1 [128,256] -> W1t bf16 [256,128]; W2 [256,128] -> W2t [128,256]
// (transposed so GEMM B-fragments are contiguous 16B loads per lane)
// ---------------------------------------------------------------------------
__global__ void convert_weights_kernel(const float* __restrict__ W1,
                                       const float* __restrict__ W2,
                                       unsigned short* __restrict__ W1t,
                                       unsigned short* __restrict__ W2t) {
    int i = blockIdx.x * blockDim.x + threadIdx.x;      // 0 .. 65535
    if (i < DIM * DIM2) {
        int k = i / DIM2, n = i % DIM2;                 // W1[k][n]
        W1t[n * DIM + k] = f2bf(W1[i]);
    } else {
        int j = i - DIM * DIM2;
        int k = j / DIM, n = j % DIM;                   // W2[k][n]
        W2t[n * DIM2 + k] = f2bf(W2[j]);
    }
}

// ---------------------------------------------------------------------------
// K2: edge scatter: out32[dst] += relu(x[src] + emb1[e0] + emb2[e1])
// 32 lanes per edge, float4 per lane, fp32 atomics (land in 192MB L2)
// ---------------------------------------------------------------------------
__global__ void edge_scatter_kernel(const float* __restrict__ x,
                                    const int*   __restrict__ edge_index,
                                    const int*   __restrict__ edge_attr,
                                    const float* __restrict__ emb1,
                                    const float* __restrict__ emb2,
                                    float* __restrict__ out32) {
    int gid = blockIdx.x * blockDim.x + threadIdx.x;
    int e = gid >> 5;
    int t = gid & 31;
    if (e >= N_EDGES) return;
    int src = edge_index[e];
    int dst = edge_index[N_EDGES + e];
    int e0 = edge_attr[2 * e];     e0 = min(max(e0, 0), 5);
    int e1 = edge_attr[2 * e + 1]; e1 = min(max(e1, 0), 2);
    int f = t * 4;
    float4 xv = *(const float4*)(x    + (size_t)src * DIM + f);
    float4 a  = *(const float4*)(emb1 + (size_t)e0  * DIM + f);
    float4 b  = *(const float4*)(emb2 + (size_t)e1  * DIM + f);
    float m0 = fmaxf(xv.x + a.x + b.x, 0.0f);
    float m1 = fmaxf(xv.y + a.y + b.y, 0.0f);
    float m2 = fmaxf(xv.z + a.z + b.z, 0.0f);
    float m3 = fmaxf(xv.w + a.w + b.w, 0.0f);
    float* o = out32 + (size_t)dst * DIM + f;
    atomicAdd(o + 0, m0);
    atomicAdd(o + 1, m1);
    atomicAdd(o + 2, m2);
    atomicAdd(o + 3, m3);
}

// ---------------------------------------------------------------------------
// K3: fp32 -> bf16 (packed 64-bit stores)
// ---------------------------------------------------------------------------
__global__ void f32_to_bf16_kernel(const float* __restrict__ in,
                                   unsigned short* __restrict__ out) {
    int i = blockIdx.x * blockDim.x + threadIdx.x;      // float4 index
    float4 v = ((const float4*)in)[i];
    unsigned long long p =
        (unsigned long long)f2bf(v.x)        |
        ((unsigned long long)f2bf(v.y) << 16) |
        ((unsigned long long)f2bf(v.z) << 32) |
        ((unsigned long long)f2bf(v.w) << 48);
    ((unsigned long long*)out)[i] = p;
}

// ---------------------------------------------------------------------------
// WMMA fragment load: row-major bf16 matrix, row `row`, K-offset k0.
// ISA 16-bit A layout (16x32): lanes 0-15 hold K {0..7, 16..23},
// lanes 16-31 hold K {8..15, 24..31}. B uses the same pattern on Bt rows.
// ---------------------------------------------------------------------------
__device__ __forceinline__ v16bf load_frag(const __bf16* __restrict__ rowp,
                                           int k0, int laneHi) {
    const __bf16* p = rowp + k0 + (laneHi ? 8 : 0);
    v8bf lo = *(const v8bf*)(p);        // K +0..7   (or +8..15)
    v8bf hi = *(const v8bf*)(p + 16);   // K +16..23 (or +24..31)
    return __builtin_shufflevector(lo, hi,
        0, 1, 2, 3, 4, 5, 6, 7, 8, 9, 10, 11, 12, 13, 14, 15);
}

// ---------------------------------------------------------------------------
// K4/K5: GEMM  C[M,N] = A[M,K] (bf16 rowmaj) * Bt[N,K]^T (bf16) + bias
// One 16x16 tile per wave, 8 waves per block. K accumulated in fp32 via
// v_wmma_f32_16x16x32_bf16. mode 0: relu -> bf16 out; mode 1: fp32 out.
// ---------------------------------------------------------------------------
__global__ void __launch_bounds__(256)
gemm_bf16_wmma_kernel(const __bf16* __restrict__ A,
                      const __bf16* __restrict__ Bt,
                      const float*  __restrict__ bias,
                      int K, int Ncols, int nTiles,
                      unsigned short* __restrict__ outBf,
                      float* __restrict__ outF32,
                      int mode) {
    int wave   = threadIdx.x >> 5;
    int lane   = threadIdx.x & 31;
    int tile   = blockIdx.x * 8 + wave;
    int mT     = tile / nTiles;
    int nT     = tile % nTiles;
    int laneLo = lane & 15;
    int laneHi = lane >> 4;

    const __bf16* arow = A  + (size_t)(mT * 16 + laneLo) * K;
    const __bf16* brow = Bt + (size_t)(nT * 16 + laneLo) * K;

    v8f c = {0.f, 0.f, 0.f, 0.f, 0.f, 0.f, 0.f, 0.f};
    for (int k0 = 0; k0 < K; k0 += 32) {
        v16bf a = load_frag(arow, k0, laneHi);
        v16bf b = load_frag(brow, k0, laneHi);
        c = __builtin_amdgcn_wmma_f32_16x16x32_bf16(
                /*neg_a=*/false, a, /*neg_b=*/false, b,
                /*c_mod=*/(short)0, c, /*reuse_a=*/false, /*reuse_b=*/false);
    }

    // D layout: VGPR r -> M = r + 8*laneHi, N = laneLo
    int nG = nT * 16 + laneLo;
    float bv = bias[nG];
#pragma unroll
    for (int r = 0; r < 8; ++r) {
        int mG = mT * 16 + r + (laneHi ? 8 : 0);
        float v = c[r] + bv;
        if (mode == 0) {
            v = fmaxf(v, 0.0f);
            outBf[(size_t)mG * Ncols + nG] = f2bf(v);
        } else {
            outF32[(size_t)mG * Ncols + nG] = v;
        }
    }
}

// ---------------------------------------------------------------------------
extern "C" void kernel_launch(void* const* d_in, const int* in_sizes, int n_in,
                              void* d_out, int out_size, void* d_ws, size_t ws_size,
                              hipStream_t stream) {
    const float* x          = (const float*)d_in[0];
    const int*   edge_index = (const int*)  d_in[1];
    const int*   edge_attr  = (const int*)  d_in[2];
    const float* emb1       = (const float*)d_in[3];
    const float* emb2       = (const float*)d_in[4];
    const float* W1         = (const float*)d_in[5];
    const float* b1         = (const float*)d_in[6];
    const float* W2         = (const float*)d_in[7];
    const float* b2         = (const float*)d_in[8];
    const float* eps        = (const float*)d_in[9];
    float* out              = (float*)d_out;

    // workspace layout (256B aligned offsets)
    char* ws = (char*)d_ws;
    float*          out32 = (float*)         (ws);              // 25,600,000 B
    unsigned short* outbf = (unsigned short*)(ws + 25600000);   // 12,800,000 B
    unsigned short* hbf   = (unsigned short*)(ws + 38400000);   // 25,600,000 B
    unsigned short* w1t   = (unsigned short*)(ws + 64000000);   //     65,536 B
    unsigned short* w2t   = (unsigned short*)(ws + 64065536);   //     65,536 B

    // K0: residual init (6.4M floats, float4 per thread)
    init_out_kernel<<<6250, 256, 0, stream>>>(x, eps, out32);
    // K1: weight convert+transpose (2 * 32768 elems)
    convert_weights_kernel<<<256, 256, 0, stream>>>(W1, W2, w1t, w2t);
    // K2: edge gather + relu + scatter-add (32 lanes/edge)
    edge_scatter_kernel<<<(N_EDGES * 32) / 256, 256, 0, stream>>>(
        x, edge_index, edge_attr, emb1, emb2, out32);
    // K3: activations -> bf16
    f32_to_bf16_kernel<<<6250, 256, 0, stream>>>(out32, outbf);
    // K4: h = relu(out @ W1 + b1)   M=50000, N=256, K=128 -> 50000 tiles
    gemm_bf16_wmma_kernel<<<6250, 256, 0, stream>>>(
        (const __bf16*)outbf, (const __bf16*)w1t, b1,
        /*K=*/DIM, /*Ncols=*/DIM2, /*nTiles=*/16, hbf, nullptr, /*mode=*/0);
    // K5: out = h @ W2 + b2         M=50000, N=128, K=256 -> 25000 tiles
    gemm_bf16_wmma_kernel<<<3125, 256, 0, stream>>>(
        (const __bf16*)hbf, (const __bf16*)w2t, b2,
        /*K=*/DIM2, /*Ncols=*/DIM, /*nTiles=*/8, nullptr, out, /*mode=*/1);
}
